// CompactBilinearPooling_53154515255355
// MI455X (gfx1250) — matrix operand order repaired and checked by reading, not compiled, not run on hardware
//
#include <hip/hip_runtime.h>

// ---------------------------------------------------------------------------
// Compact bilinear pooling on MI455X (gfx1250), wave32 + WMMA + TDM.
//
//   out[p,:] = unnormalized-IDFT( rfft(y1_p) * rfft(y2_p) )          (real)
//   y1,y2    = count-sketch scatters of the two 512-channel pixel vectors.
//
// 8192-pt FFTs as four-step (64x128) GEMM-FFTs on v_wmma_f32_16x16x32_f16
// (f16 DFT operands, f32 accumulate). One packed complex forward FFT per
// pixel (y1 + i*y2), Hermitian separation, and out = Re(DFT(conj(F1*F2)))
// reuses the same GEMM path with the imaginary half of stage 3 elided.
// DFT/twiddle tables are built once in global memory (resident in the 192MB
// L2) and DMA'd into LDS per workgroup with tensor_load_to_lds, overlapped
// with the count-sketch scatter (waited via s_wait_tensorcnt).
// ---------------------------------------------------------------------------

#define D_DIM 8192
#define C_DIM 512
#define HW    196
#define B_DIM 16
#define P_DIM (B_DIM * HW)   // 3136 pixels
#define TPB   256            // 8 wave32 waves

typedef __attribute__((ext_vector_type(16))) _Float16 v16h;
typedef __attribute__((ext_vector_type(8)))  float    v8f;
typedef __attribute__((ext_vector_type(4)))  unsigned int u32x4;
typedef __attribute__((ext_vector_type(8)))  int      i32x8;
typedef __attribute__((ext_vector_type(4)))  int      i32x4;

#define TWO_PI 6.283185307179586f

// Shared table blob layout (identical in global memory and at LDS offset 0)
#define OFF_W64R   0        //  64x64  f16
#define OFF_W64I   8192
#define OFF_W128R  16384    // 128x128 f16
#define OFF_W128I  49152
#define OFF_TR     81920    //  64x128 f32 step-2 twiddle
#define OFF_TI     114688
#define WBLOB_BYTES 147456  // = 36864 f32 elements

__device__ __forceinline__ v8f wmma_f16(v16h a, v16h b, v8f c) {
  return __builtin_amdgcn_wmma_f32_16x16x32_f16(false, a, false, b, (short)0, c,
                                                false, false);
}

// ---- A fragment (16x32 of row-major [M][ld]): lanes 0-15 rows, K chunks
// {kb..kb+7, kb+16..kb+23}, kb = 0 / 8 by lane half.
__device__ __forceinline__ v16h fragA_h(const _Float16* src, int m0, int k0,
                                        int ld, int lane, bool neg) {
  const _Float16* p = src + (m0 + (lane & 15)) * ld + k0 + ((lane & 16) ? 8 : 0);
  v16h f;
#pragma unroll
  for (int h = 0; h < 8; ++h) {
    _Float16 a = p[h], b = p[16 + h];
    f[h]     = neg ? (_Float16)(-a) : a;
    f[8 + h] = neg ? (_Float16)(-b) : b;
  }
  return f;
}
__device__ __forceinline__ v16h fragA_f(const float* src, int m0, int k0,
                                        int ld, int lane) {
  const float* p = src + (m0 + (lane & 15)) * ld + k0 + ((lane & 16) ? 8 : 0);
  v16h f;
#pragma unroll
  for (int h = 0; h < 8; ++h) {
    f[h]     = (_Float16)p[h];
    f[8 + h] = (_Float16)p[16 + h];
  }
  return f;
}

// ---- B fragment (32x16 of B[K][N]) from transposed storage BT[N][ld]:
// lane -> column N, K chunks {kb..kb+7, kb+16..kb+23}, kb = 0 / 8 by lane half.
__device__ __forceinline__ v16h fragB_h(const _Float16* srcT, int n0, int k0,
                                        int ld, int lane, bool neg) {
  const _Float16* p = srcT + (n0 + (lane & 15)) * ld + k0 + ((lane & 16) ? 8 : 0);
  v16h f;
#pragma unroll
  for (int h = 0; h < 16; ++h) {
    _Float16 a = p[h + (h >= 8 ? 8 : 0)];
    f[h] = neg ? (_Float16)(-a) : a;
  }
  return f;
}
__device__ __forceinline__ v16h fragB_f(const float* srcT, int n0, int k0,
                                        int ld, int lane) {
  const float* p = srcT + (n0 + (lane & 15)) * ld + k0 + ((lane & 16) ? 8 : 0);
  v16h f;
#pragma unroll
  for (int h = 0; h < 16; ++h) f[h] = (_Float16)p[h + (h >= 8 ? 8 : 0)];
  return f;
}

// ---- C/D 16x16 f32: VGPR r of lane L holds (M = r + 8*(L>=16), N = L%16)
__device__ __forceinline__ void storeC(float* dst, v8f acc, int m0, int n0,
                                       int sm, int sn, int lane) {
  int n  = n0 + (lane & 15);
  int mb = m0 + ((lane & 16) ? 8 : 0);
#pragma unroll
  for (int r = 0; r < 8; ++r) dst[(mb + r) * sm + n * sn] = acc[r];
}

// ---------------------------------------------------------------------------
// 8192-pt complex DFT, four-step N = 64*128, on WMMA.
//   srcT planes: srcT[n2*64 + n1] = x[128*n1 + n2]   (transposed input)
//   zr/zi      : scratch planes row-major [64][128]  (k1 x n2)
//   dst planes : indexed by k = k1 + 64*k2 ; IMAG=false -> real output only
// ---------------------------------------------------------------------------
template <bool IMAG>
__device__ __forceinline__ void dft8192(
    const float* srcTr, const float* srcTi, float* zr, float* zi,
    float* dstr, float* dsti, const _Float16* w64r, const _Float16* w64i,
    const _Float16* w128r, const _Float16* w128i, const float* tr,
    const float* ti, int tid) {
  const int lane = tid & 31;
  const int wv   = tid >> 5;   // 0..7

  // step 1: Z[k1][n2] = sum_n1 W64[k1][n1] * u[128*n1 + n2]
  {
    const int nt = wv;
    v16h ur0 = fragB_f(srcTr, nt * 16, 0, 64, lane);
    v16h ur1 = fragB_f(srcTr, nt * 16, 32, 64, lane);
    v16h ui0 = fragB_f(srcTi, nt * 16, 0, 64, lane);
    v16h ui1 = fragB_f(srcTi, nt * 16, 32, 64, lane);
#pragma unroll
    for (int mt = 0; mt < 4; ++mt) {
      v8f ar = {}; v8f ai = {};
      v16h wr0  = fragA_h(w64r, mt * 16, 0, 64, lane, false);
      v16h wi0  = fragA_h(w64i, mt * 16, 0, 64, lane, false);
      v16h nwi0 = fragA_h(w64i, mt * 16, 0, 64, lane, true);
      ar = wmma_f16(wr0, ur0, ar); ar = wmma_f16(nwi0, ui0, ar);
      ai = wmma_f16(wr0, ui0, ai); ai = wmma_f16(wi0, ur0, ai);
      v16h wr1  = fragA_h(w64r, mt * 16, 32, 64, lane, false);
      v16h wi1  = fragA_h(w64i, mt * 16, 32, 64, lane, false);
      v16h nwi1 = fragA_h(w64i, mt * 16, 32, 64, lane, true);
      ar = wmma_f16(wr1, ur1, ar); ar = wmma_f16(nwi1, ui1, ar);
      ai = wmma_f16(wr1, ui1, ai); ai = wmma_f16(wi1, ur1, ai);
      storeC(zr, ar, mt * 16, nt * 16, 128, 1, lane);
      storeC(zi, ai, mt * 16, nt * 16, 128, 1, lane);
    }
  }
  __syncthreads();

  // step 2: Z'[k1][n2] = Z * e^{-2*pi*i*k1*n2/8192}  (table from LDS)
  for (int i = tid; i < 64 * 128; i += TPB) {
    float a = zr[i], b = zi[i];
    float cs = tr[i], sn = ti[i];
    zr[i] = a * cs - b * sn;
    zi[i] = a * sn + b * cs;
  }
  __syncthreads();

  // step 3: X[k1 + 64*k2] = sum_n2 Z'[k1][n2] * W128[n2][k2]  (W128 symmetric)
  {
    const int nt = wv;   // k2 tile
    v16h br[4], bi[4], nbi[4];
#pragma unroll
    for (int kk = 0; kk < 4; ++kk) {
      br[kk]  = fragB_h(w128r, nt * 16, kk * 32, 128, lane, false);
      bi[kk]  = fragB_h(w128i, nt * 16, kk * 32, 128, lane, false);
      nbi[kk] = fragB_h(w128i, nt * 16, kk * 32, 128, lane, true);
    }
#pragma unroll
    for (int mt = 0; mt < 4; ++mt) {
      v8f xr = {}; v8f xi = {};
#pragma unroll
      for (int kk = 0; kk < 4; ++kk) {
        v16h azr = fragA_f(zr, mt * 16, kk * 32, 128, lane);
        v16h azi = fragA_f(zi, mt * 16, kk * 32, 128, lane);
        xr = wmma_f16(azr, br[kk], xr);
        xr = wmma_f16(azi, nbi[kk], xr);
        if (IMAG) {
          xi = wmma_f16(azr, bi[kk], xi);
          xi = wmma_f16(azi, br[kk], xi);
        }
      }
      storeC(dstr, xr, mt * 16, nt * 16, 1, 64, lane);   // k = k1 + 64*k2
      if (IMAG) storeC(dsti, xi, mt * 16, nt * 16, 1, 64, lane);
    }
  }
  __syncthreads();
}

// ---------------------------------------------------------------------------
// Init: build the DFT / twiddle tables once in global memory (L2-resident).
// ---------------------------------------------------------------------------
__global__ void wtab_init(char* __restrict__ blob) {
  int t = blockIdx.x * TPB + threadIdx.x;     // 16384 threads
  _Float16* w64r  = (_Float16*)(blob + OFF_W64R);
  _Float16* w64i  = (_Float16*)(blob + OFF_W64I);
  _Float16* w128r = (_Float16*)(blob + OFF_W128R);
  _Float16* w128i = (_Float16*)(blob + OFF_W128I);
  float*    tr    = (float*)(blob + OFF_TR);
  float*    ti    = (float*)(blob + OFF_TI);
  if (t < 64 * 64) {
    int r = t >> 6, c = t & 63;
    float sn, cs;
    __sincosf(-(float)(r * c) * (TWO_PI / 64.0f), &sn, &cs);
    w64r[t] = (_Float16)cs; w64i[t] = (_Float16)sn;
  }
  if (t < 128 * 128) {
    int r = t >> 7, c = t & 127;
    float sn, cs;
    __sincosf(-(float)(r * c) * (TWO_PI / 128.0f), &sn, &cs);
    w128r[t] = (_Float16)cs; w128i[t] = (_Float16)sn;
  }
  if (t < 64 * 128) {
    int k1 = t >> 7, n2 = t & 127;
    float sn, cs;
    __sincosf(-(float)(k1 * n2) * (TWO_PI / 8192.0f), &sn, &cs);
    tr[t] = cs; ti[t] = sn;
  }
}

// ---------------------------------------------------------------------------
// Prep: recover (h[c], s[c]) from dense count-sketch (one nonzero per row).
// ---------------------------------------------------------------------------
__global__ void sketch_extract(const float* __restrict__ S1,
                               const float* __restrict__ S2,
                               int* __restrict__ h, float* __restrict__ s) {
  int m = blockIdx.x >> 9;
  int r = blockIdx.x & 511;
  const float* row = (m ? S2 : S1) + (size_t)r * D_DIM;
  int*   hrow = h + m * C_DIM;
  float* srow = s + m * C_DIM;
  for (int c = threadIdx.x; c < D_DIM; c += TPB) {
    float v = row[c];
    if (v != 0.0f) { hrow[r] = c; srow[r] = v; }
  }
}

// ---------------------------------------------------------------------------
// Main: one workgroup (8 waves) per pixel.
// ---------------------------------------------------------------------------
__global__ void __launch_bounds__(TPB, 1)
cbp_main(const float* __restrict__ x1, const float* __restrict__ x2,
         const int* __restrict__ h1, const float* __restrict__ s1,
         const int* __restrict__ h2, const float* __restrict__ s2,
         const char* __restrict__ wblob, float* __restrict__ out) {
  extern __shared__ char smem[];
  _Float16* w64r  = (_Float16*)(smem + OFF_W64R);
  _Float16* w64i  = (_Float16*)(smem + OFF_W64I);
  _Float16* w128r = (_Float16*)(smem + OFF_W128R);
  _Float16* w128i = (_Float16*)(smem + OFF_W128I);
  float*    tr    = (float*)(smem + OFF_TR);
  float*    ti    = (float*)(smem + OFF_TI);
  float* bufAr = (float*)(smem + WBLOB_BYTES);   // 4 x 8192-f32 planes
  float* bufAi = bufAr + D_DIM;
  float* bufBr = bufAi + D_DIM;
  float* bufBi = bufBr + D_DIM;

  const int tid = threadIdx.x;
  const int p   = blockIdx.x;
  const int b   = p / HW;
  const int hw  = p - b * HW;

  // ---- TDM: DMA the 144KB table blob (L2-resident) into LDS offset 0.
  // Issued by wave 0 only; overlaps with the zeroing + scatter below.
  if (tid < 32) {
    unsigned long long ga = (unsigned long long)(const void*)wblob;
    u32x4 g0;
    g0[0] = 1u;                                          // count=1, user desc
    g0[1] = 0u;                                          // lds_addr = 0
    g0[2] = (unsigned)ga;                                // global_addr lo
    g0[3] = (unsigned)((ga >> 32) & 0x01FFFFFFull)       // global_addr hi
          | (2u << 30);                                  // type = 2 (image)
    const unsigned ne = WBLOB_BYTES / 4;                 // 36864 f32 elements
    i32x8 g1;
    g1[0] = (int)(2u << 16);                 // data_size = 4B
    g1[1] = (int)((ne & 0xFFFFu) << 16);     // tensor_dim0[15:0]
    g1[2] = (int)((ne >> 16) | (1u << 16));  // tensor_dim0[31:16] | tensor_dim1=1
    g1[3] = (int)((ne & 0xFFFFu) << 16);     // tile_dim0 = ne
    g1[4] = 1;                               // tile_dim1 = 1
    g1[5] = (int)ne;                         // tensor_dim0_stride[31:0]
    g1[6] = 0;
    g1[7] = 0;
    i32x4 gz4 = {0, 0, 0, 0};                // groups 2/3 unused (<=2D tensor)
    i32x8 gz8 = {0, 0, 0, 0, 0, 0, 0, 0};    // trailing group (6-arg variant)
    __builtin_amdgcn_tensor_load_to_lds(g0, g1, gz4, gz4, gz8, 0);
  }

  // ---- zero scatter planes, then count-sketch scatter u = y1 + i*y2 into
  // transposed layout: bin d -> plane[(d%128)*64 + d/128]
  for (int i = tid; i < 2 * D_DIM; i += TPB) bufAr[i] = 0.0f;
  __syncthreads();
#pragma unroll
  for (int c0 = 0; c0 < C_DIM; c0 += TPB) {
    int c  = c0 + tid;
    int d1 = h1[c];
    atomicAdd(&bufAr[(d1 & 127) * 64 + (d1 >> 7)],
              s1[c] * x1[(size_t)(b * C_DIM + c) * HW + hw]);
    int d2 = h2[c];
    atomicAdd(&bufAi[(d2 & 127) * 64 + (d2 >> 7)],
              s2[c] * x2[(size_t)(b * C_DIM + c) * HW + hw]);
  }
  __builtin_amdgcn_s_wait_tensorcnt(0);   // tables landed (no-op for waves 1-7)
  __syncthreads();

  // ---- forward FFT of packed input -> bufA (indexed by k)
  dft8192<true>(bufAr, bufAi, bufBr, bufBi, bufAr, bufAi,
                w64r, w64i, w128r, w128i, tr, ti, tid);

  // ---- Hermitian separation + pointwise product: G = conj(F1*F2),
  // written transposed into bufB (input layout of the second DFT).
  for (int k = tid; k < 4096; k += TPB) {
    int j = (D_DIM - k) & (D_DIM - 1);
    float arv = bufAr[k], aiv = bufAi[k];
    float brv = bufAr[j], biv = bufAi[j];
    float f1r = 0.5f * (arv + brv), f1i = 0.5f * (aiv - biv);
    float f2r = 0.5f * (aiv + biv), f2i = 0.5f * (brv - arv);
    float hr = f1r * f2r - f1i * f2i;
    float hi = f1r * f2i + f1i * f2r;
    int tk = (k & 127) * 64 + (k >> 7);
    int tj = (j & 127) * 64 + (j >> 7);
    bufBr[tk] = hr; bufBi[tk] = -hi;    // G[k]   = conj(H[k])
    bufBr[tj] = hr; bufBi[tj] = hi;     // G[N-k] = H[k]
  }
  if (tid == 0) {                        // Nyquist bin
    float arv = bufAr[4096], aiv = bufAi[4096];
    int tk = (4096 & 127) * 64 + (4096 >> 7);
    bufBr[tk] = arv * aiv; bufBi[tk] = 0.0f;
  }
  __syncthreads();

  // ---- out = Re( DFT(G) ) == unnormalized IDFT of H (real part only)
  dft8192<false>(bufBr, bufBi, bufAr, bufAi, bufBr, bufBr,
                 w64r, w64i, w128r, w128i, tr, ti, tid);

  float* dst = out + (size_t)p * D_DIM;
  for (int m = tid; m < D_DIM; m += TPB) dst[m] = bufBr[m];
}

// ---------------------------------------------------------------------------
extern "C" void kernel_launch(void* const* d_in, const int* in_sizes, int n_in,
                              void* d_out, int out_size, void* d_ws,
                              size_t ws_size, hipStream_t stream) {
  const float* x1 = (const float*)d_in[0];
  const float* x2 = (const float*)d_in[1];
  const float* S1 = (const float*)d_in[2];
  const float* S2 = (const float*)d_in[3];

  int*   h    = (int*)d_ws;                 // h1[512], h2[512]
  float* s    = (float*)d_ws + 2 * C_DIM;   // s1[512], s2[512]
  char*  blob = (char*)d_ws + 8192;         // 144KB table blob (256B aligned)

  wtab_init<<<64, TPB, 0, stream>>>(blob);
  sketch_extract<<<2 * C_DIM, TPB, 0, stream>>>(S1, S2, h, s);

  constexpr size_t SMEM = WBLOB_BYTES + (size_t)4 * D_DIM * sizeof(float);
  static_assert(SMEM == 278528, "272KB LDS <= 320KB/WGP");
  (void)hipFuncSetAttribute((const void*)cbp_main,
                            hipFuncAttributeMaxDynamicSharedMemorySize,
                            (int)SMEM);

  cbp_main<<<P_DIM, TPB, SMEM, stream>>>(x1, x2, h, s, h + C_DIM, s + C_DIM,
                                         blob, (float*)d_out);
}
